// Attention_5798205849797
// MI455X (gfx1250) — matrix operand order repaired and checked by reading, not compiled, not run on hardware
//
#include <hip/hip_runtime.h>
#include <hip/hip_bf16.h>

// ---------------------------------------------------------------------------
// Attention (B=4, S=4096, D=128, single head, softmax over unmasked scores).
// Kernel 1: QKV projection, bf16 WMMA; Q,K row-major bf16, V *transposed*
//           (vT[b][d][s]) into workspace so flash staging is pure 2D copies.
// Kernel 2: flash attention with double-buffered ASYNC global->LDS staging
//           (GLOBAL_LOAD_ASYNC_TO_LDS_B128 + s_wait_asynccnt) and bf16 WMMA.
// ---------------------------------------------------------------------------

typedef __attribute__((ext_vector_type(16))) __bf16 v16bf;
typedef __attribute__((ext_vector_type(8)))  float  v8f;
typedef __attribute__((ext_vector_type(4)))  int    v4i;

#define AS1 __attribute__((address_space(1)))
#define AS3 __attribute__((address_space(3)))

union FragAB {           // 16 bf16 = 8 VGPRs = one WMMA A/B operand
    uint4  u[2];
    v16bf  v;
};

__device__ __forceinline__ v8f wmma_bf16(v16bf a, v16bf b, v8f c) {
    return __builtin_amdgcn_wmma_f32_16x16x32_bf16(false, a, false, b,
                                                   (short)0, c, false, false);
}

static constexpr int Dk   = 128;     // embedding dim
static constexpr int SEQ  = 4096;    // sequence length
static constexpr int NTOK = 4 * SEQ; // B*S = 16384
static constexpr int LSTR = 136;     // padded bf16 row stride (272B = 17*16B)
static constexpr int VSTR = 40;      // padded key stride for vT tile (80B)

// ---- async global->LDS copy of one 16-byte chunk (per-lane addresses) ------
#if __has_builtin(__builtin_amdgcn_global_load_async_to_lds_b128)
#define USE_ASYNC 1
#else
#define USE_ASYNC 0
#endif

__device__ __forceinline__ void copy16(const __bf16* g, __bf16* l) {
#if USE_ASYNC
    __builtin_amdgcn_global_load_async_to_lds_b128(
        (AS1 v4i*)(unsigned long long)(const void*)g,
        (AS3 v4i*)(unsigned int)(unsigned long long)(const void*)l,
        0, 0);
#else
    *(uint4*)l = *(const uint4*)g;
#endif
}

#if USE_ASYNC
#if __has_builtin(__builtin_amdgcn_s_wait_asynccnt)
#define WAIT_ASYNC(n) __builtin_amdgcn_s_wait_asynccnt(n)
#else
#define WAIT_ASYNC(n) asm volatile("s_wait_asynccnt %0" :: "i"(n) : "memory")
#endif
#else
#define WAIT_ASYNC(n) do {} while (0)
#endif

// ---------------------------------------------------------------------------
// Kernel 1: projections.  grid=(256,3), block=128 (4 waves).
// Each wave computes a 16x128 output tile: 8 N-tiles x 4 K-steps of WMMA.
// proj 0/1 (Q,K): row-major bf16.  proj 2 (V): transposed vT[b][d][s].
// ---------------------------------------------------------------------------
__global__ void __launch_bounds__(128)
qkv_project_kernel(const float* __restrict__ x,
                   const float* __restrict__ wq,
                   const float* __restrict__ wk,
                   const float* __restrict__ wv,
                   __bf16* __restrict__ qkv_base)
{
    const int proj = blockIdx.y;
    const float* w = (proj == 0) ? wq : (proj == 1) ? wk : wv;
    __bf16* out = qkv_base + (size_t)proj * NTOK * Dk;

    const int row_base = blockIdx.x * 64;          // 64 token rows per block

    __shared__ alignas(16) __bf16 wt[Dk * LSTR];   // W transposed: wt[n][k]
    __shared__ alignas(16) __bf16 xs[64 * LSTR];   // x tile row-major bf16

    const int tid = threadIdx.x;

    for (int idx = tid; idx < Dk * Dk; idx += 128) {   // W^T, fp32->bf16
        int k = idx >> 7, n = idx & 127;
        wt[n * LSTR + k] = (__bf16)w[idx];
    }
    const float* xrow = x + (size_t)row_base * Dk;
    for (int idx = tid; idx < 64 * Dk; idx += 128) {   // x tile
        xs[(idx >> 7) * LSTR + (idx & 127)] = (__bf16)xrow[idx];
    }
    __syncthreads();

    const int lane = tid & 31;
    const int wid  = tid >> 5;
    const int m    = lane & 15;
    const int h    = lane >> 4;

    FragAB a[4];   // A 16x32: half h holds K = h*8..+7 and 16+h*8..+7
#pragma unroll
    for (int kt = 0; kt < 4; ++kt) {
        const __bf16* p = &xs[(wid * 16 + m) * LSTR + kt * 32 + h * 8];
        a[kt].u[0] = *(const uint4*)p;
        a[kt].u[1] = *(const uint4*)(p + 16);
    }

    v8f accv[8];
#pragma unroll
    for (int nt = 0; nt < 8; ++nt) {
        v8f acc = {};
#pragma unroll
        for (int kt = 0; kt < 4; ++kt) {
            FragAB bfrag;  // B 32x16: lane = column n, half h holds K = h*8..+7 and 16+h*8..+7
            const __bf16* p = &wt[(nt * 16 + m) * LSTR + kt * 32 + h * 8];
            bfrag.u[0] = *(const uint4*)p;
            bfrag.u[1] = *(const uint4*)(p + 16);
            acc = wmma_bf16(a[kt].v, bfrag.v, acc);
        }
        accv[nt] = acc;
    }

    // ---- epilogue: branch ONCE (block-uniform), straight-line stores -------
    const int rw = row_base + wid * 16;            // wave's first output row
    if (proj != 2) {                               // Q, K: row-major [s][d]
#pragma unroll
        for (int nt = 0; nt < 8; ++nt)
#pragma unroll
            for (int r = 0; r < 8; ++r)
                out[(size_t)(rw + r + 8 * h) * Dk + nt * 16 + m] =
                    (__bf16)accv[nt][r];
    } else {                                       // V: transposed [b][d][s]
        const int bb = row_base >> 12;             // whole block in one batch
        __bf16* vout = out + (size_t)bb * Dk * SEQ - (size_t)bb * SEQ;
#pragma unroll
        for (int nt = 0; nt < 8; ++nt)
#pragma unroll
            for (int r = 0; r < 8; ++r)
                vout[(size_t)(nt * 16 + m) * SEQ + (rw + r + 8 * h)] =
                    (__bf16)accv[nt][r];
    }
}

// ---------------------------------------------------------------------------
// Kernel 2: flash attention. grid=256 (4 batches x 64 q-blocks), block=128.
// Double-buffered async staging of K (row-major) and vT (pre-transposed).
// ---------------------------------------------------------------------------
__global__ void __launch_bounds__(128)
flash_attn_kernel(const __bf16* __restrict__ qg,
                  const __bf16* __restrict__ kg,
                  const __bf16* __restrict__ vTg_base,
                  float* __restrict__ outg)
{
    constexpr float SCALE = 0.08838834764831845f;  // 1/sqrt(128)

    const int b  = blockIdx.x >> 6;
    const int qb = blockIdx.x & 63;
    const __bf16* q   = qg  + (size_t)b * SEQ * Dk;
    const __bf16* kp  = kg  + (size_t)b * SEQ * Dk;
    const __bf16* vTg = vTg_base + (size_t)b * Dk * SEQ;  // vT[d][s]
    float* ob = outg + (size_t)b * SEQ * Dk;

    __shared__ alignas(16) __bf16 klds[2][32 * LSTR];  // K tiles, row-major
    __shared__ alignas(16) __bf16 vt[2][Dk * VSTR];    // vT tiles: vt[d][key]
    __shared__ alignas(16) __bf16 ps[4 * 16 * 32];     // per-wave P scratch

    const int tid  = threadIdx.x;
    const int lane = tid & 31;
    const int wid  = tid >> 5;
    const int m    = lane & 15;
    const int h    = lane >> 4;
    const int row0 = qb * 64 + wid * 16;

    // Q A-fragments, loaded once
    FragAB aq[4];
#pragma unroll
    for (int kt = 0; kt < 4; ++kt) {
        const __bf16* p = q + (size_t)(row0 + m) * Dk + kt * 32 + h * 8;
        aq[kt].u[0] = *(const uint4*)p;
        aq[kt].u[1] = *(const uint4*)(p + 16);
    }

    float mi[8], li[8];
    v8f acc[8];
#pragma unroll
    for (int r = 0; r < 8; ++r) { mi[r] = -1e30f; li[r] = 0.0f; }
#pragma unroll
    for (int nt = 0; nt < 8; ++nt) acc[nt] = (v8f){};

    // ---- tile stage: 8 b128 copies per thread (4 K-chunks + 4 V-chunks) ----
    auto stage_tile = [&](int buf, int kb) {
#pragma unroll
        for (int i = 0; i < 4; ++i) {                  // K: 32x128 bf16
            int c = tid + i * 128;                     // 512 chunks of 8 elems
            int key = c >> 4, d0 = (c & 15) * 8;
            copy16(&kp[(size_t)(kb + key) * Dk + d0], &klds[buf][key * LSTR + d0]);
        }
#pragma unroll
        for (int i = 0; i < 4; ++i) {                  // vT: 128x32 bf16
            int c = tid + i * 128;
            int d = c >> 2, key0 = (c & 3) * 8;
            copy16(&vTg[(size_t)d * SEQ + kb + key0], &vt[buf][d * VSTR + key0]);
        }
    };

    constexpr int NTILES = SEQ / 32;
    stage_tile(0, 0);                                   // prologue: tile 0

    for (int it = 0; it < NTILES; ++it) {
        const int cur = it & 1;
        __syncthreads();                  // spare buffer's readers are done
        if (it + 1 < NTILES) {
            stage_tile(1 - cur, (it + 1) * 32);         // prefetch next tile
            WAIT_ASYNC(8);                              // tile `it` landed
        } else {
            WAIT_ASYNC(0);
        }
        __syncthreads();                  // tile `it` visible to all waves

        // ---- scores: S0 = Q x K[0:16]^T, S1 = Q x K[16:32]^T ----
        v8f s0 = (v8f){}, s1 = (v8f){};
#pragma unroll
        for (int kt = 0; kt < 4; ++kt) {
            FragAB b0, b1;
            const __bf16* p0 = &klds[cur][m * LSTR + kt * 32 + h * 8];
            b0.u[0] = *(const uint4*)p0;  b0.u[1] = *(const uint4*)(p0 + 16);
            const __bf16* p1 = &klds[cur][(16 + m) * LSTR + kt * 32 + h * 8];
            b1.u[0] = *(const uint4*)p1;  b1.u[1] = *(const uint4*)(p1 + 16);
            s0 = wmma_bf16(aq[kt].v, b0.v, s0);
            s1 = wmma_bf16(aq[kt].v, b1.v, s1);
        }

        // ---- online softmax over the 32 new columns ----
        __bf16* pw = &ps[wid * 512];
#pragma unroll
        for (int r = 0; r < 8; ++r) {
            float x0 = s0[r] * SCALE;
            float x1 = s1[r] * SCALE;
            float vmax = fmaxf(x0, x1);
#pragma unroll
            for (int off = 8; off >= 1; off >>= 1)
                vmax = fmaxf(vmax, __shfl_xor(vmax, off, 16));
            float mnew  = fmaxf(mi[r], vmax);
            float alpha = __expf(mi[r] - mnew);
            float p0 = __expf(x0 - mnew);
            float p1 = __expf(x1 - mnew);
            float rs = p0 + p1;
#pragma unroll
            for (int off = 8; off >= 1; off >>= 1)
                rs += __shfl_xor(rs, off, 16);
            li[r] = li[r] * alpha + rs;
            mi[r] = mnew;
#pragma unroll
            for (int nt = 0; nt < 8; ++nt) acc[nt][r] *= alpha;
            pw[(r + 8 * h) * 32 + m]      = (__bf16)p0;   // C -> row-major 16x32
            pw[(r + 8 * h) * 32 + 16 + m] = (__bf16)p1;
        }

        // ---- reload P in A-matrix layout (wave-private LDS round trip) ----
        FragAB ap;
        {
            const __bf16* p = &pw[m * 32 + h * 8];
            ap.u[0] = *(const uint4*)p;
            ap.u[1] = *(const uint4*)(p + 16);
        }

        // ---- O += P x V ----
#pragma unroll
        for (int nt = 0; nt < 8; ++nt) {
            FragAB bv;   // B 32x16: lane = out column d, K = key (contig in vt)
            const __bf16* p = &vt[cur][(nt * 16 + m) * VSTR + h * 8];
            bv.u[0] = *(const uint4*)p;
            bv.u[1] = *(const uint4*)(p + 16);
            acc[nt] = wmma_bf16(ap.v, bv.v, acc[nt]);
        }
    }

    // ---- epilogue ----
#pragma unroll
    for (int nt = 0; nt < 8; ++nt) {
#pragma unroll
        for (int r = 0; r < 8; ++r) {
            ob[(size_t)(row0 + r + 8 * h) * Dk + nt * 16 + m] =
                acc[nt][r] / li[r];
        }
    }
}

// ---------------------------------------------------------------------------
// Host launcher
// ---------------------------------------------------------------------------
extern "C" void kernel_launch(void* const* d_in, const int* in_sizes, int n_in,
                              void* d_out, int out_size, void* d_ws, size_t ws_size,
                              hipStream_t stream) {
    const float* x  = (const float*)d_in[0];
    // d_in[1] = token_attention_masks (unused: reference softmaxes unmasked scores)
    const float* wq = (const float*)d_in[2];
    const float* wk = (const float*)d_in[3];
    const float* wv = (const float*)d_in[4];

    __bf16* qkv = (__bf16*)d_ws;                 // Q | K | vT, 4 MB each
    const size_t one = (size_t)NTOK * Dk;

    qkv_project_kernel<<<dim3(256, 3), 128, 0, stream>>>(x, wq, wk, wv, qkv);
    flash_attn_kernel<<<dim3(256), 128, 0, stream>>>(qkv, qkv + one, qkv + 2 * one,
                                                     (float*)d_out);
}
